// Attention_80522046865889
// MI455X (gfx1250) — matrix operand order repaired and checked by reading, not compiled, not run on hardware
//
#include <hip/hip_runtime.h>

// CDNA5 / gfx1250 flash attention (causal), fp32 in/out, f16 WMMA compute.
// B=2, L=2048, H=16, E=64. Double-buffered LDS K/V tiles, 1 barrier/tile,
// register-staged global loads overlap WMMA+softmax, exp2-domain softmax,
// DPP row_ror butterflies for reductions, batched WMMA operand fragment loads.

typedef __attribute__((ext_vector_type(16))) _Float16 v16h;
typedef __attribute__((ext_vector_type(8)))  float    v8f;

#define B_    2
#define L_    2048
#define H_    16
#define E_    64
#define ROWS  (H_ * E_)      // stride between consecutive sequence positions
#define QTILE 128            // query rows per workgroup (8 waves x 16 rows)
#define KTILE 32             // key rows per iteration (= WMMA K for f16)
#define NWAVE 8

// DPP row rotate within the 16-lane row (wave32: row = 16 lanes).
template <int CTRL>
__device__ __forceinline__ float dppmove(float v) {
    return __builtin_bit_cast(float,
        __builtin_amdgcn_update_dpp(0, __builtin_bit_cast(int, v),
                                    CTRL, 0xF, 0xF, true));
}
__device__ __forceinline__ float rowmax16(float v) {
    v = fmaxf(v, dppmove<0x121>(v));   // row_ror:1
    v = fmaxf(v, dppmove<0x122>(v));   // row_ror:2
    v = fmaxf(v, dppmove<0x124>(v));   // row_ror:4
    v = fmaxf(v, dppmove<0x128>(v));   // row_ror:8
    return v;
}
__device__ __forceinline__ float rowsum16(float v) {
    v += dppmove<0x121>(v);
    v += dppmove<0x122>(v);
    v += dppmove<0x124>(v);
    v += dppmove<0x128>(v);
    return v;
}

__launch_bounds__(256)
__global__ void fa_causal_wmma(const float* __restrict__ Q,
                               const float* __restrict__ K,
                               const float* __restrict__ V,
                               float* __restrict__ O) {
    __shared__ _Float16 Kl[2][KTILE][E_ + 8];
    __shared__ _Float16 Vt[2][E_][KTILE + 8];
    __shared__ _Float16 Pl[NWAVE][16][KTILE];

    const int tid  = threadIdx.x;
    const int wave = tid >> 5;
    const int lane = tid & 31;
    const int ln   = lane & 15;
    const int hi   = lane >> 4;

    const int nQT = L_ / QTILE;
    const int qt  = blockIdx.x % nQT;
    const int bh  = blockIdx.x / nQT;
    const int b   = bh / H_;
    const int h   = bh % H_;

    const int qTileBase = qt * QTILE;
    const int qw0       = qTileBase + wave * 16;

    const size_t headOff = (size_t)b * L_ * ROWS + (size_t)h * E_;
    const float* Qp = Q + headOff;
    const float* Kp = K + headOff;
    const float* Vp = V + headOff;
    float*       Op = O + headOff;

    // Q fragment (A-layout), 1/sqrt(E) * log2(e) folded so exp2 is used directly.
    const float QSCALE = 0.125f * 1.4426950408889634f;
    v16h qf[2];
    {
        const float* qrow = Qp + (size_t)(qw0 + ln) * ROWS;
        #pragma unroll
        for (int s = 0; s < 2; ++s)
            #pragma unroll
            for (int e = 0; e < 16; ++e) {
                const int k = 16 * hi + e;
                qf[s][e] = (_Float16)(qrow[s * 32 + k] * QSCALE);
            }
    }

    v8f oacc[4];
    #pragma unroll
    for (int et = 0; et < 4; ++et) oacc[et] = (v8f){};
    float mrow[8], lacc[8];
    #pragma unroll
    for (int r = 0; r < 8; ++r) { mrow[r] = -__builtin_inff(); lacc[r] = 0.0f; }

    const int nkt  = (qTileBase + QTILE) / KTILE;
    const int srow = tid >> 3;
    const int se0  = (tid & 7) * 8;

    float kreg[8], vreg[8];
    auto loadTile = [&](int kBase) {    // global -> registers (overlaps compute)
        const float* kr = Kp + (size_t)(kBase + srow) * ROWS + se0;
        const float* vr = Vp + (size_t)(kBase + srow) * ROWS + se0;
        #pragma unroll
        for (int j = 0; j < 8; ++j) { kreg[j] = kr[j]; vreg[j] = vr[j]; }
        if (kBase + KTILE < nkt * KTILE) {   // global_prefetch_b8 of tile t+2
            __builtin_prefetch(kr + KTILE * ROWS, 0, 3);
            __builtin_prefetch(vr + KTILE * ROWS, 0, 3);
        }
    };
    auto storeTile = [&](int buf) {     // registers -> LDS (f16, K packed pairs)
        #pragma unroll
        for (int j = 0; j < 4; ++j) {
            auto pk = __builtin_amdgcn_cvt_pkrtz(kreg[2 * j], kreg[2 * j + 1]);
            *(decltype(pk)*)&Kl[buf][srow][se0 + 2 * j] = pk;
        }
        #pragma unroll
        for (int j = 0; j < 8; ++j)
            Vt[buf][se0 + j][srow] = (_Float16)vreg[j];   // transpose into LDS
    };

    loadTile(0);
    storeTile(0);
    __syncthreads();

    for (int kt = 0; kt < nkt; ++kt) {
        const int kBase = kt * KTILE;
        const int buf   = kt & 1;
        if (kt + 1 < nkt) loadTile(kBase + KTILE);   // issue next tile's loads

        if (kBase <= qw0 + 15) {                      // wave-uniform causal gate
            // ---- S = Q * K^T : batch all 4 B-fragments, then WMMA chain ----
            v16h kb[4];                                // [s-slice * 2 + key-subtile]
            #pragma unroll
            for (int s = 0; s < 2; ++s)
                #pragma unroll
                for (int t = 0; t < 2; ++t)
                    #pragma unroll
                    for (int e = 0; e < 16; ++e)
                        kb[s * 2 + t][e] = Kl[buf][t * 16 + ln][s * 32 + 16 * hi + e];

            v8f s0 = (v8f){}, s1 = (v8f){};
            s0 = __builtin_amdgcn_wmma_f32_16x16x32_f16(false, qf[0], false, kb[0],
                                                        (short)0, s0, false, false);
            s1 = __builtin_amdgcn_wmma_f32_16x16x32_f16(false, qf[0], false, kb[1],
                                                        (short)0, s1, false, false);
            s0 = __builtin_amdgcn_wmma_f32_16x16x32_f16(false, qf[1], false, kb[2],
                                                        (short)0, s0, false, false);
            s1 = __builtin_amdgcn_wmma_f32_16x16x32_f16(false, qf[1], false, kb[3],
                                                        (short)0, s1, false, false);

            // ---- online softmax (per-lane partial sums; DPP row-max) ----
            auto softmax = [&](bool masked) {
                #pragma unroll
                for (int r = 0; r < 8; ++r) {
                    float a = s0[r], c = s1[r];
                    if (masked) {
                        const int qq = qw0 + r + 8 * hi;
                        if (kBase + ln      > qq) a = -__builtin_inff();
                        if (kBase + 16 + ln > qq) c = -__builtin_inff();
                    }
                    const float mx = rowmax16(fmaxf(a, c));
                    const float mn = fmaxf(mrow[r], mx);
                    const float al = __builtin_amdgcn_exp2f(mrow[r] - mn);
                    const float p0 = __builtin_amdgcn_exp2f(a - mn);
                    const float p1 = __builtin_amdgcn_exp2f(c - mn);
                    lacc[r] = lacc[r] * al + (p0 + p1);
                    mrow[r] = mn;
                    oacc[0][r] *= al; oacc[1][r] *= al;
                    oacc[2][r] *= al; oacc[3][r] *= al;
                    Pl[wave][r + 8 * hi][ln]      = (_Float16)p0;
                    Pl[wave][r + 8 * hi][16 + ln] = (_Float16)p1;
                }
            };
            if (kBase + KTILE - 1 <= qw0) softmax(false);   // steady state: no masks
            else                          softmax(true);    // 1-2 boundary tiles

            // ---- P (A-layout via LDS) * V : batch A + 4 B-fragments ----
            v16h pa, vb[4];
            #pragma unroll
            for (int e = 0; e < 16; ++e) {
                const int k = 16 * hi + e;
                pa[e] = Pl[wave][ln][k];
            }
            #pragma unroll
            for (int et = 0; et < 4; ++et)
                #pragma unroll
                for (int e = 0; e < 16; ++e)
                    vb[et][e] = Vt[buf][et * 16 + ln][16 * hi + e];
            #pragma unroll
            for (int et = 0; et < 4; ++et)
                oacc[et] = __builtin_amdgcn_wmma_f32_16x16x32_f16(false, pa, false, vb[et],
                                                                  (short)0, oacc[et],
                                                                  false, false);
        }

        if (kt + 1 < nkt) storeTile(buf ^ 1);   // fill other buffer for next iter
        __syncthreads();                        // single barrier per tile
    }

    // ---- final row-sum reduce, normalize, store ----
    #pragma unroll
    for (int r = 0; r < 8; ++r) {
        const float inv = __builtin_amdgcn_rcpf(rowsum16(lacc[r]));
        float* orow = Op + (size_t)(qw0 + r + 8 * hi) * ROWS;
        #pragma unroll
        for (int et = 0; et < 4; ++et)
            orow[et * 16 + ln] = oacc[et][r] * inv;
    }
}

extern "C" void kernel_launch(void* const* d_in, const int* in_sizes, int n_in,
                              void* d_out, int out_size, void* d_ws, size_t ws_size,
                              hipStream_t stream) {
    (void)in_sizes; (void)n_in; (void)d_ws; (void)ws_size; (void)out_size;
    const float* Q = (const float*)d_in[0];
    const float* K = (const float*)d_in[1];
    const float* V = (const float*)d_in[2];
    float*       O = (float*)d_out;
    const int grid = B_ * H_ * (L_ / QTILE);   // 512 workgroups
    fa_causal_wmma<<<grid, 256, 0, stream>>>(Q, K, V, O);
}
